// GraphConvEncoder_6150393168617
// MI455X (gfx1250) — compile-verified, matrix operand order checked by reading, not run on hardware
//
#include <hip/hip_runtime.h>
#include <hip/hip_bf16.h>

// GraphConv encoder for MI455X (gfx1250), wave32 + WMMA bf16.
//
// Pipeline (all on `stream`):
//   0. zero degree accumulators
//   1. edge degree count (f32 atomics) -> 2. ns/nd = rsqrt(deg)
//   3. pack W1/W2/Wc0/Wc1 (f32 -> bf16, WMMA-B fragment order)
//   4. MLP1: [N,256]x[256,512] WMMA + bias + LayerNorm + ReLU
//            -> H1 stored in WMMA-A fragment order (bf16)
//   5. MLP2: [N,512]x[512,128] WMMA + bias, scaled by ns -> hs (f32)
//   6/8. zero agg; scatter: agg[dst] += hs[src]   (wave-per-edge, f32 atomics,
//        node state is 51MB -> L2-resident on the 192MB L2)
//   7/9. conv GEMM: relu((agg*nd) @ Wc + bc), optionally *ns -> hs / d_out

typedef __attribute__((ext_vector_type(16))) __bf16 v16bf;
typedef __attribute__((ext_vector_type(8)))  float  v8f;

#define N_NODES 100000
#define N_EDGES 1600000

union AFrag { v16bf v; unsigned int u[8]; };

__device__ __forceinline__ v8f vzero8() {
  v8f r;
#pragma unroll
  for (int i = 0; i < 8; ++i) r[i] = 0.0f;
  return r;
}

// WMMA 16x16x32 bf16 -> f32
__device__ __forceinline__ v8f wmma_bf16(v16bf a, v16bf b, v8f c) {
  return __builtin_amdgcn_wmma_f32_16x16x32_bf16(false, a, false, b,
                                                 (short)0, c, false, false);
}

// A-fragment load from a row-major bf16 LDS tile (row stride `stride` elems).
// Per ISA 16-bit A 16x32 layout: lane = m + 16*g ; vgpr v holds K pair
//   k0 = kc + ((v&4)<<2) + (g<<3) + ((v&3)<<1)
__device__ __forceinline__ v16bf load_a_frag_lds(const __bf16* lds, int stride,
                                                 int kc, int lane) {
  int m = lane & 15;
  int g = (lane >> 4) & 1;
  AFrag f;
#pragma unroll
  for (int vv = 0; vv < 8; ++vv) {
    int k0 = kc + ((vv & 4) << 2) + (g << 3) + ((vv & 3) << 1);
    f.u[vv] = *(const unsigned int*)(lds + m * stride + k0);
  }
  return f.v;
}

// ---------------- utility kernels ----------------

__global__ void zero_kernel(float* __restrict__ p, long n) {
  long i = (long)blockIdx.x * blockDim.x + threadIdx.x;
  long stride = (long)gridDim.x * blockDim.x;
  for (; i < n; i += stride) p[i] = 0.0f;
}

// Pack f32 weight matrix [K x Ncol] into WMMA-B fragment order (bf16):
// fragment (kt, nt): per-lane 16 contiguous bf16 at
//   out[((kt*ntiles + nt)*32 + lane)*16 + j]
__global__ void pack_b_kernel(const float* __restrict__ W, __bf16* __restrict__ out,
                              int K, int Ncol) {
  int t = blockIdx.x * blockDim.x + threadIdx.x;
  int lane = t & 31;
  int frag = t >> 5;
  int ntiles = Ncol >> 4;
  int ktiles = K >> 5;
  if (frag >= ktiles * ntiles) return;
  int kt = frag / ntiles;
  int nt = frag - kt * ntiles;
  int n = nt * 16 + (lane & 15);
  int g = (lane >> 4) & 1;
  v16bf frg;
#pragma unroll
  for (int vv = 0; vv < 8; ++vv) {
    int k = kt * 32 + ((vv & 4) << 2) + (g << 3) + ((vv & 3) << 1);
    frg[2 * vv]     = (__bf16)W[(size_t)k * Ncol + n];
    frg[2 * vv + 1] = (__bf16)W[(size_t)(k + 1) * Ncol + n];
  }
  *(v16bf*)(out + ((size_t)frag * 32 + lane) * 16) = frg;
}

__global__ void degree_kernel(const int* __restrict__ src, const int* __restrict__ dst,
                              float* __restrict__ dout, float* __restrict__ din, int E) {
  int e = blockIdx.x * blockDim.x + threadIdx.x;
  if (e < E) {
    atomicAdd(&dout[src[e]], 1.0f);
    atomicAdd(&din[dst[e]], 1.0f);
  }
}

__global__ void norm_kernel(const float* __restrict__ dout, const float* __restrict__ din,
                            float* __restrict__ nsv, float* __restrict__ ndv, int n) {
  int i = blockIdx.x * blockDim.x + threadIdx.x;
  if (i < n) {
    float a = dout[i];
    float b = din[i];
    nsv[i] = (a > 0.0f) ? rsqrtf(a) : 0.0f;
    ndv[i] = (b > 0.0f) ? rsqrtf(b) : 0.0f;
  }
}

// wave-per-edge scatter: agg[dst] += hs[src] over F=128 (4 floats / lane)
__global__ void scatter_kernel(const float* __restrict__ hs,
                               const int* __restrict__ src, const int* __restrict__ dst,
                               float* __restrict__ agg, int E) {
  int t = blockIdx.x * blockDim.x + threadIdx.x;
  int e = t >> 5;
  int lane = t & 31;
  if (e >= E) return;
  int s = src[e];
  int d = dst[e];
  const float4 v = *(const float4*)(hs + (size_t)s * 128 + lane * 4);
  float* p = agg + (size_t)d * 128 + lane * 4;
  atomicAdd(p + 0, v.x);
  atomicAdd(p + 1, v.y);
  atomicAdd(p + 2, v.z);
  atomicAdd(p + 3, v.w);
}

// ---------------- MLP stage 1: concat -> GEMM(256->512) -> +b1 -> LN -> ReLU ----------------
// block = 256 threads (8 waves), computes a 16-node x 512 tile. Output H1 is
// written in WMMA-A fragment order (bf16) for the next GEMM.
__global__ __launch_bounds__(256) void mlp1_kernel(
    const float* __restrict__ features, const int* __restrict__ labels,
    const float* __restrict__ label_embed,
    const __bf16* __restrict__ W1p, const float* __restrict__ b1,
    const float* __restrict__ g1, const float* __restrict__ be1,
    __bf16* __restrict__ H1p) {
  __shared__ alignas(16) __bf16 sA[16 * 258];   // 16 x 256, pad 2
  __shared__ float sC[16 * 516];                // 16 x 512, pad 4
  __shared__ float sRed[2][16][16];
  __shared__ float sMu[16], sRs[16];

  const int tid = threadIdx.x;
  const int bt = blockIdx.x;

  // stage A = [features | label_embed[labels]] (f32 -> bf16)
  {
    int m = tid >> 4;
    int c0 = (tid & 15) * 16;
    int node = bt * 16 + m;
    int lab = labels[node];
    const float* srcp = (c0 < 128) ? (features + (size_t)node * 128 + c0)
                                   : (label_embed + (size_t)lab * 128 + (c0 - 128));
#pragma unroll
    for (int i = 0; i < 16; ++i)
      sA[m * 258 + c0 + i] = (__bf16)srcp[i];
  }
  __syncthreads();

  const int w = tid >> 5;       // wave -> cols [w*64, w*64+64)
  const int lane = tid & 31;
  const int g = lane >> 4;

  v8f acc[4];
#pragma unroll
  for (int t = 0; t < 4; ++t) acc[t] = vzero8();

#pragma unroll
  for (int kt = 0; kt < 8; ++kt) {
    v16bf a = load_a_frag_lds(sA, 258, kt * 32, lane);
#pragma unroll
    for (int t = 0; t < 4; ++t) {
      int nt = w * 4 + t;
      v16bf b = *(const v16bf*)(W1p + (((size_t)kt * 32 + nt) * 32 + lane) * 16);
      acc[t] = wmma_bf16(a, b, acc[t]);
    }
  }

  // bias + park f32 tile in LDS
#pragma unroll
  for (int t = 0; t < 4; ++t) {
    int n = w * 64 + t * 16 + (lane & 15);
    float bn = b1[n];
#pragma unroll
    for (int r = 0; r < 8; ++r) {
      int m2 = r + 8 * g;
      sC[m2 * 516 + n] = acc[t][r] + bn;
    }
  }
  __syncthreads();

  // LayerNorm over 512 per row (16 threads x 32 cols per row)
  {
    int r = tid & 15;
    int ch = tid >> 4;
    float s = 0.0f, s2 = 0.0f;
#pragma unroll
    for (int i = 0; i < 32; ++i) {
      float x = sC[r * 516 + ch * 32 + i];
      s += x;
      s2 += x * x;
    }
    sRed[0][r][ch] = s;
    sRed[1][r][ch] = s2;
  }
  __syncthreads();
  if (tid < 16) {
    float s = 0.0f, s2 = 0.0f;
#pragma unroll
    for (int i = 0; i < 16; ++i) { s += sRed[0][tid][i]; s2 += sRed[1][tid][i]; }
    float mu = s * (1.0f / 512.0f);
    float var = s2 * (1.0f / 512.0f) - mu * mu;
    sMu[tid] = mu;
    sRs[tid] = rsqrtf(var + 1e-5f);
  }
  __syncthreads();
  {
    int r = tid & 15;
    int ch = tid >> 4;
    float mu = sMu[r], rs = sRs[r];
#pragma unroll
    for (int i = 0; i < 32; ++i) {
      int c = ch * 32 + i;
      float x = sC[r * 516 + c];
      float y = (x - mu) * rs * g1[c] + be1[c];
      sC[r * 516 + c] = fmaxf(y, 0.0f);
    }
  }
  __syncthreads();

  // pack to global in WMMA-A fragment order (K=512 -> 16 ktiles per node-tile)
  for (int fl = tid; fl < 512; fl += 256) {
    int kt = fl >> 5;
    int ln = fl & 31;
    int m3 = ln & 15;
    int g2 = ln >> 4;
    v16bf frg;
#pragma unroll
    for (int j = 0; j < 16; ++j) {
      int kk = (j & 7) + 8 * g2 + ((j & 8) << 1);
      frg[j] = (__bf16)sC[m3 * 516 + kt * 32 + kk];
    }
    *(v16bf*)(H1p + ((size_t)(bt * 16 + kt) * 32 + ln) * 16) = frg;
  }
}

// ---------------- MLP stage 2: GEMM(512->128) + b2, *ns -> hs (f32) ----------------
__global__ __launch_bounds__(256) void mlp2_kernel(
    const __bf16* __restrict__ H1p, const __bf16* __restrict__ W2p,
    const float* __restrict__ b2, const float* __restrict__ nsv,
    float* __restrict__ hs) {
  const int tid = threadIdx.x;
  const int bt = blockIdx.x;
  const int w = tid >> 5;      // ntile 0..7
  const int lane = tid & 31;
  const int g = lane >> 4;

  v8f acc = vzero8();
#pragma unroll
  for (int kt = 0; kt < 16; ++kt) {
    v16bf a = *(const v16bf*)(H1p + ((size_t)(bt * 16 + kt) * 32 + lane) * 16);
    v16bf b = *(const v16bf*)(W2p + ((size_t)(kt * 8 + w) * 32 + lane) * 16);
    acc = wmma_bf16(a, b, acc);
  }
  int n = w * 16 + (lane & 15);
  float bn = b2[n];
#pragma unroll
  for (int r = 0; r < 8; ++r) {
    int node = bt * 16 + r + 8 * g;
    hs[(size_t)node * 128 + n] = (acc[r] + bn) * nsv[node];
  }
}

// ---------------- GraphConv transform: relu((agg*nd) @ Wc + bc) [* ns] ----------------
__global__ __launch_bounds__(256) void conv_kernel(
    const float* __restrict__ agg, const float* __restrict__ ndv,
    const float* __restrict__ nsv,   // null -> no output scaling
    const __bf16* __restrict__ Wp, const float* __restrict__ bias,
    float* __restrict__ out) {
  __shared__ alignas(16) __bf16 sA[16 * 130];   // 16 x 128, pad 2
  const int tid = threadIdx.x;
  const int bt = blockIdx.x;
  {
    int m = tid >> 4;
    int c0 = (tid & 15) * 8;
    int node = bt * 16 + m;
    float nm = ndv[node];
    const float* p = agg + (size_t)node * 128 + c0;
#pragma unroll
    for (int i = 0; i < 8; ++i)
      sA[m * 130 + c0 + i] = (__bf16)(p[i] * nm);
  }
  __syncthreads();

  const int w = tid >> 5;
  const int lane = tid & 31;
  const int g = lane >> 4;
  v8f acc = vzero8();
#pragma unroll
  for (int kt = 0; kt < 4; ++kt) {
    v16bf a = load_a_frag_lds(sA, 130, kt * 32, lane);
    v16bf b = *(const v16bf*)(Wp + ((size_t)(kt * 8 + w) * 32 + lane) * 16);
    acc = wmma_bf16(a, b, acc);
  }
  int n = w * 16 + (lane & 15);
  float bn = bias[n];
#pragma unroll
  for (int r = 0; r < 8; ++r) {
    int node = bt * 16 + r + 8 * g;
    float v = fmaxf(acc[r] + bn, 0.0f);
    if (nsv) v *= nsv[node];
    out[(size_t)node * 128 + n] = v;
  }
}

// ---------------- launch ----------------

extern "C" void kernel_launch(void* const* d_in, const int* in_sizes, int n_in,
                              void* d_out, int out_size, void* d_ws, size_t ws_size,
                              hipStream_t stream) {
  (void)in_sizes; (void)n_in; (void)out_size; (void)ws_size;

  const float* features    = (const float*)d_in[0];
  const int*   labels      = (const int*)d_in[1];
  const int*   edge_src    = (const int*)d_in[2];
  const int*   edge_dst    = (const int*)d_in[3];
  const float* label_embed = (const float*)d_in[4];
  const float* W1  = (const float*)d_in[5];
  const float* b1  = (const float*)d_in[6];
  const float* g1  = (const float*)d_in[7];
  const float* be1 = (const float*)d_in[8];
  const float* W2  = (const float*)d_in[9];
  const float* b2  = (const float*)d_in[10];
  const float* Wc0 = (const float*)d_in[11];
  const float* bc0 = (const float*)d_in[12];
  const float* Wc1 = (const float*)d_in[13];
  const float* bc1 = (const float*)d_in[14];
  float* out = (float*)d_out;

  char* ws = (char*)d_ws;
  size_t cur = 0;
  auto alloc = [&](size_t bytes) -> void* {
    void* p = ws + cur;
    cur = (cur + bytes + 255) & ~(size_t)255;
    return p;
  };

  __bf16* W1p  = (__bf16*)alloc((size_t)256 * 512 * 2);
  __bf16* W2p  = (__bf16*)alloc((size_t)512 * 128 * 2);
  __bf16* Wc0p = (__bf16*)alloc((size_t)128 * 128 * 2);
  __bf16* Wc1p = (__bf16*)alloc((size_t)128 * 128 * 2);
  float* deg_out = (float*)alloc((size_t)N_NODES * 4);
  float* deg_in  = (float*)alloc((size_t)N_NODES * 4);
  float* nsv = (float*)alloc((size_t)N_NODES * 4);
  float* ndv = (float*)alloc((size_t)N_NODES * 4);
  __bf16* H1p = (__bf16*)alloc((size_t)N_NODES * 512 * 2);
  float* hs  = (float*)alloc((size_t)N_NODES * 128 * 4);
  float* agg = (float*)alloc((size_t)N_NODES * 128 * 4);

  const int nblk = N_NODES / 16;          // 6250 (N divisible by 16)
  const int eblk = (N_EDGES + 255) / 256; // edge-parallel blocks

  // degrees + norms
  zero_kernel<<<512, 256, 0, stream>>>(deg_out, N_NODES);
  zero_kernel<<<512, 256, 0, stream>>>(deg_in, N_NODES);
  degree_kernel<<<eblk, 256, 0, stream>>>(edge_src, edge_dst, deg_out, deg_in, N_EDGES);
  norm_kernel<<<(N_NODES + 255) / 256, 256, 0, stream>>>(deg_out, deg_in, nsv, ndv, N_NODES);

  // pack weights into WMMA-B fragment order (bf16)
  pack_b_kernel<<<(8 * 32 * 32 + 255) / 256, 256, 0, stream>>>(W1, W1p, 256, 512);
  pack_b_kernel<<<(16 * 8 * 32 + 255) / 256, 256, 0, stream>>>(W2, W2p, 512, 128);
  pack_b_kernel<<<(4 * 8 * 32 + 255) / 256, 256, 0, stream>>>(Wc0, Wc0p, 128, 128);
  pack_b_kernel<<<(4 * 8 * 32 + 255) / 256, 256, 0, stream>>>(Wc1, Wc1p, 128, 128);

  // MLP
  mlp1_kernel<<<nblk, 256, 0, stream>>>(features, labels, label_embed, W1p, b1, g1, be1, H1p);
  mlp2_kernel<<<nblk, 256, 0, stream>>>(H1p, W2p, b2, nsv, hs);

  // GraphConv layer 0
  zero_kernel<<<2048, 256, 0, stream>>>(agg, (long)N_NODES * 128);
  scatter_kernel<<<N_EDGES / 8, 256, 0, stream>>>(hs, edge_src, edge_dst, agg, N_EDGES);
  conv_kernel<<<nblk, 256, 0, stream>>>(agg, ndv, nsv, Wc0p, bc0, hs);

  // GraphConv layer 1 (final -> d_out, no ns scaling)
  zero_kernel<<<2048, 256, 0, stream>>>(agg, (long)N_NODES * 128);
  scatter_kernel<<<N_EDGES / 8, 256, 0, stream>>>(hs, edge_src, edge_dst, agg, N_EDGES);
  conv_kernel<<<nblk, 256, 0, stream>>>(agg, ndv, nullptr, Wc1p, bc1, out);
}
